// AttnBlock_71004399337864
// MI455X (gfx1250) — compile-verified, matrix operand order checked by reading, not run on hardware
//
#include <hip/hip_runtime.h>
#include <hip/hip_bf16.h>

// AttnBlock fused pipeline for MI455X (gfx1250, wave32, WMMA f16->f32).
// Workspace requirement: ~44 MB (weights f16 + hT/qT/kT/v/o f16 buffers).

#define DEVFN static __device__ __forceinline__

typedef _Float16 half_t;
typedef __attribute__((ext_vector_type(16))) _Float16 v16h;
typedef __attribute__((ext_vector_type(8)))  _Float16 v8h;
typedef __attribute__((ext_vector_type(8)))  float    v8f;

constexpr int Cc = 512;    // channels
constexpr int Nn = 4096;   // H*W tokens
constexpr int Bb = 2;      // batch
constexpr int TQ = 32;     // query tile (attention)
constexpr int TK = 64;     // key tile (attention)

// Per-lane fragment load for 16-bit WMMA operands.
// Lane L holds 16 f16 values: two contiguous 8-element chunks at
// base (K = h*8..h*8+7) and base+16 (K = 16+h*8..), h = L>>4, per the
// CDNA5 16-bit A/B VGPR striping.  Caller passes rowptr + kbase + h*8.
DEVFN v16h load_frag(const half_t* p) {
    v8h lo = *(const v8h*)(p);
    v8h hi = *(const v8h*)(p + 16);
    return __builtin_shufflevector(lo, hi,
        0,1,2,3,4,5,6,7,8,9,10,11,12,13,14,15);
}

DEVFN v8f wmma16(v16h a, v16h b, v8f c) {
    // D = A(16x32 f16) * B(32x16 f16) + C(16x16 f32)
    return __builtin_amdgcn_wmma_f32_16x16x32_f16(
        /*neg_a=*/false, a, /*neg_b=*/false, b,
        /*c_mod=*/(short)0, c, /*reuse_a=*/false, /*reuse_b=*/false);
}

DEVFN v8f vzero() {
    v8f z = {0.f,0.f,0.f,0.f,0.f,0.f,0.f,0.f};
    return z;
}

// CDNA5 async global->LDS copy (ASYNCcnt-tracked, CDNA5 ISA).
// lds_off = byte offset within LDS (low 32 bits of generic shared address).
DEVFN void async_copy_b128(unsigned lds_off, const half_t* gptr) {
    asm volatile("global_load_async_to_lds_b128 %0, %1, off"
                 :: "v"(lds_off), "v"(gptr) : "memory");
}
DEVFN void wait_async0() {
    asm volatile("s_wait_asynccnt 0x0" ::: "memory");
}

// ---------------------------------------------------------------- weights->f16
__global__ void k_wcvt(const float* wq, const float* wk, const float* wv,
                       const float* wo, half_t* dst) {
    const float* src = (blockIdx.y == 0) ? wq :
                       (blockIdx.y == 1) ? wk :
                       (blockIdx.y == 2) ? wv : wo;
    size_t i = (size_t)blockIdx.x * 256 + threadIdx.x;
    dst[(size_t)blockIdx.y * Cc * Cc + i] = (half_t)src[i];
}

// ---------------------------------------------------------------- GroupNorm
// One block per (batch, group): 16 channels x 4096 tokens.
// Writes normalized activations as hT[b][n][c] in f16.
__global__ void k_gn(const float* __restrict__ x,
                     const float* __restrict__ gs,
                     const float* __restrict__ gb,
                     half_t* __restrict__ hT) {
    const int bg = blockIdx.x;
    const int b  = bg >> 5;          // 32 groups
    const int g  = bg & 31;
    const int c0 = g * 16;
    const float* xp = x + (size_t)b * Cc * Nn;

    float s = 0.f, s2 = 0.f;
    for (int i = threadIdx.x; i < 16 * Nn; i += 256) {
        const int ci = i >> 12;          // /4096
        const int n  = i & (Nn - 1);
        float v = xp[(size_t)(c0 + ci) * Nn + n];
        s += v; s2 += v * v;
    }
    __shared__ float rs[256], rq[256];
    rs[threadIdx.x] = s; rq[threadIdx.x] = s2;
    __syncthreads();
    for (int off = 128; off > 0; off >>= 1) {
        if (threadIdx.x < off) {
            rs[threadIdx.x] += rs[threadIdx.x + off];
            rq[threadIdx.x] += rq[threadIdx.x + off];
        }
        __syncthreads();
    }
    __shared__ float smean, srstd;
    if (threadIdx.x == 0) {
        const float inv = 1.0f / (16.0f * Nn);
        float mean = rs[0] * inv;
        float var  = rq[0] * inv - mean * mean;
        smean = mean;
        srstd = rsqrtf(var + 1e-6f);
    }
    __syncthreads();
    const float mean = smean, rstd = srstd;
    for (int i = threadIdx.x; i < 16 * Nn; i += 256) {
        const int ci = i >> 12;
        const int n  = i & (Nn - 1);
        const int c  = c0 + ci;
        float v = (xp[(size_t)c * Nn + n] - mean) * rstd * gs[c] + gb[c];
        hT[((size_t)b * Nn + n) * Cc + c] = (half_t)v;
    }
}

// ---------------------------------------------------------------- QKV GEMM
// out[co,n] = W[co,:] . h[:,n] + bias.  Block tile 128co x 64n, 8 waves,
// each wave 32co x 32n (2 A-frags x 2 B-frags -> 4 WMMA per k-step).
// mode 0 -> qT[b][n][c] (pre-scaled by C^-1/2), 1 -> kT[b][n][c], 2 -> v[b][c][n]
__global__ void k_qkv(const half_t* __restrict__ w16,
                      const half_t* __restrict__ hT,
                      const float* __restrict__ bq,
                      const float* __restrict__ bk,
                      const float* __restrict__ bv,
                      half_t* __restrict__ qT,
                      half_t* __restrict__ kT,
                      half_t* __restrict__ vB) {
    const int n0   = blockIdx.x * 64;
    const int co0  = blockIdx.y * 128;
    const int b    = blockIdx.z / 3;
    const int mode = blockIdx.z % 3;
    const int lane = threadIdx.x & 31, wid = threadIdx.x >> 5;
    const int lm = lane & 15, lh = lane >> 4;
    const int cs = wid & 3;      // 32-row co subtile
    const int ns = wid >> 2;     // 0..1 -> 32 n columns

    const half_t* W    = w16 + (size_t)mode * Cc * Cc;
    const float*  bias = (mode == 0) ? bq : (mode == 1) ? bk : bv;

    const half_t* wrow0 = W  + (size_t)(co0 + cs * 32 + lm) * Cc;
    const half_t* wrow1 = wrow0 + (size_t)16 * Cc;
    const half_t* hrow0 = hT + ((size_t)b * Nn + n0 + ns * 32 + lm) * Cc;
    const half_t* hrow1 = hrow0 + (size_t)16 * Cc;

    v8f acc[2][2];
    acc[0][0] = vzero(); acc[0][1] = vzero();
    acc[1][0] = vzero(); acc[1][1] = vzero();

    for (int kc = 0; kc < Cc; kc += 32) {
        v16h a0 = load_frag(wrow0 + kc + lh * 8);
        v16h a1 = load_frag(wrow1 + kc + lh * 8);
        v16h b0 = load_frag(hrow0 + kc + lh * 8);
        v16h b1 = load_frag(hrow1 + kc + lh * 8);
        acc[0][0] = wmma16(a0, b0, acc[0][0]);
        acc[0][1] = wmma16(a0, b1, acc[0][1]);
        acc[1][0] = wmma16(a1, b0, acc[1][0]);
        acc[1][1] = wmma16(a1, b1, acc[1][1]);
    }

    const float scale = (mode == 0) ? rsqrtf((float)Cc) : 1.0f;
    #pragma unroll
    for (int i = 0; i < 2; ++i) {
        #pragma unroll
        for (int j = 0; j < 2; ++j) {
            #pragma unroll
            for (int r = 0; r < 8; ++r) {
                const int co = co0 + cs * 32 + i * 16 + r + lh * 8; // D-frag M
                const int na = n0 + ns * 32 + j * 16 + lm;         // D-frag N
                const float v = (acc[i][j][r] + bias[co]) * scale;
                if (mode == 0) {
                    qT[((size_t)b * Nn + na) * Cc + co] = (half_t)v;
                } else if (mode == 1) {
                    kT[((size_t)b * Nn + na) * Cc + co] = (half_t)v;
                } else {
                    vB[((size_t)b * Cc + co) * Nn + na] = (half_t)v;
                }
            }
        }
    }
}

// ---------------------------------------------------------------- Flash attention
// Block: 32 queries, 8 waves.  Q tile staged once into LDS via CDNA5 async
// global->LDS copies.  Per 64-key tile:
//   phase 1: wave w -> S subtile (qi=w&1, kj=w>>1), 16 WMMA over K=C,
//            A-frags from LDS (ds_load_b128), B-frags streamed from L2.
//   phase 2: online softmax (running max/sum, rescale alpha) in LDS.
//   phase 3: wave w -> O rows qi*16, channels [cg*128,+128), 16 WMMA.
__global__ void k_attn(const half_t* __restrict__ qT,
                       const half_t* __restrict__ kT,
                       const half_t* __restrict__ vB,
                       half_t* __restrict__ oB) {
    const int b  = blockIdx.y;
    const int q0 = blockIdx.x * TQ;
    const int tid = threadIdx.x, lane = tid & 31, wid = tid >> 5;
    const int lm = lane & 15, lh = lane >> 4;
    const int qi = wid & 1;         // query subtile (both phases)
    const int kj = wid >> 1;        // key subtile (phase 1), 0..3
    const int cg = wid >> 1;        // channel group (phase 3), 0..3

    __shared__ __align__(16) half_t sQ[TQ][Cc];   // 32 KB, loaded once
    __shared__ __align__(16) float  sS[TQ][TK];
    __shared__ __align__(16) half_t sP[TQ][TK];
    __shared__ float sMax[TQ], sSum[TQ], sAlpha[TQ];
    __shared__ float sRed[TQ][8];

    if (tid < TQ) { sMax[tid] = -1e30f; sSum[tid] = 0.f; }

    // ---- stage Q tile into LDS with async copies (ASYNCcnt path)
    {
        const half_t* qbase = qT + ((size_t)b * Nn + q0) * Cc;
        for (int i = tid; i < TQ * Cc / 8; i += 256) {
            const int row  = i >> 6;          // Cc/8 = 64 chunks per row
            const int col8 = i & 63;
            unsigned lds_off = (unsigned)(uintptr_t)&sQ[row][col8 * 8];
            async_copy_b128(lds_off, qbase + (size_t)row * Cc + col8 * 8);
        }
        wait_async0();
    }

    v8f acc[8];
    #pragma unroll
    for (int t = 0; t < 8; ++t) acc[t] = vzero();

    __syncthreads();

    const half_t* qlds = &sQ[qi * 16 + lm][0];

    for (int k0 = 0; k0 < Nn; k0 += TK) {
        // ---- phase 1: S = Q . K^T subtile (f32 accum)
        const half_t* krow = kT + ((size_t)b * Nn + k0 + kj * 16 + lm) * Cc;
        __builtin_prefetch(kT + ((size_t)b * Nn + k0 + TK + kj * 16 + lm) * Cc, 0, 0);
        v8f s = vzero();
        for (int kc = 0; kc < Cc; kc += 32) {
            v16h a  = load_frag(qlds + kc + lh * 8);   // ds_load_b128 x2
            v16h bb = load_frag(krow + kc + lh * 8);
            s = wmma16(a, bb, s);
        }
        #pragma unroll
        for (int r = 0; r < 8; ++r)
            sS[qi * 16 + r + lh * 8][kj * 16 + lm] = s[r];
        __syncthreads();

        // ---- phase 2: online softmax, 8 threads per row
        {
            const int row = tid >> 3, t8 = tid & 7;
            float mloc = -1e30f;
            #pragma unroll
            for (int j = 0; j < 8; ++j) mloc = fmaxf(mloc, sS[row][t8 * 8 + j]);
            sRed[row][t8] = mloc;
            __syncthreads();
            if (t8 == 0) {
                float mnew = sMax[row];
                #pragma unroll
                for (int j = 0; j < 8; ++j) mnew = fmaxf(mnew, sRed[row][j]);
                float alpha = __expf(sMax[row] - mnew);
                sAlpha[row] = alpha;
                sMax[row]   = mnew;
                sSum[row]  *= alpha;
            }
            __syncthreads();
            const float mnew = sMax[row];
            float ps = 0.f;
            #pragma unroll
            for (int j = 0; j < 8; ++j) {
                float p = __expf(sS[row][t8 * 8 + j] - mnew);
                sP[row][t8 * 8 + j] = (half_t)p;
                ps += p;
            }
            sRed[row][t8] = ps;
            __syncthreads();
            if (t8 == 0) {
                float add = 0.f;
                #pragma unroll
                for (int j = 0; j < 8; ++j) add += sRed[row][j];
                sSum[row] += add;
            }
            __syncthreads();   // sP/sAlpha ready for phase 3
        }

        // ---- phase 3: rescale accumulators, O += P . V
        float arow[8];
        #pragma unroll
        for (int r = 0; r < 8; ++r) arow[r] = sAlpha[qi * 16 + r + lh * 8];
        #pragma unroll
        for (int t = 0; t < 8; ++t)
            #pragma unroll
            for (int r = 0; r < 8; ++r) acc[t][r] *= arow[r];

        #pragma unroll
        for (int km = 0; km < TK; km += 32) {
            v16h ap = load_frag(&sP[qi * 16 + lm][km + lh * 8]);
            #pragma unroll
            for (int t = 0; t < 8; ++t) {
                const half_t* vrow = vB + ((size_t)b * Cc + cg * 128 + t * 16 + lm) * Nn
                                        + k0 + km + lh * 8;
                v16h bv = load_frag(vrow);
                acc[t] = wmma16(ap, bv, acc[t]);
            }
        }
        __syncthreads();   // protect sS/sP/sAlpha before next tile
    }

    // ---- finalize: divide by row sums, store O[b][n][c] f16
    float rsum[8];
    #pragma unroll
    for (int r = 0; r < 8; ++r)
        rsum[r] = 1.0f / sSum[qi * 16 + r + lh * 8];
    #pragma unroll
    for (int t = 0; t < 8; ++t) {
        #pragma unroll
        for (int r = 0; r < 8; ++r) {
            const int n = q0 + qi * 16 + r + lh * 8;
            const int c = cg * 128 + t * 16 + lm;
            oB[((size_t)b * Nn + n) * Cc + c] = (half_t)(acc[t][r] * rsum[r]);
        }
    }
}

// ---------------------------------------------------------------- proj + residual
// out[b][co][n] = x + Wo[co,:] . O[:,n] + bo[co]
// Block tile 64n x 128co, each wave 32n x 32co (4 WMMA per k-step).
__global__ void k_proj(const half_t* __restrict__ wo16,
                       const half_t* __restrict__ oB,
                       const float* __restrict__ x,
                       const float* __restrict__ bo,
                       float* __restrict__ out) {
    const int n0  = blockIdx.x * 64;
    const int co0 = blockIdx.y * 128;
    const int b   = blockIdx.z;
    const int lane = threadIdx.x & 31, wid = threadIdx.x >> 5;
    const int lm = lane & 15, lh = lane >> 4;
    const int ns = wid & 1;      // 0..1 -> 32 n rows (GEMM M)
    const int cs = wid >> 1;     // 0..3 -> 32 co cols

    const half_t* orow0 = oB + ((size_t)b * Nn + n0 + ns * 32 + lm) * Cc;
    const half_t* orow1 = orow0 + (size_t)16 * Cc;
    const half_t* wrow0 = wo16 + (size_t)(co0 + cs * 32 + lm) * Cc;
    const half_t* wrow1 = wrow0 + (size_t)16 * Cc;

    v8f acc[2][2];
    acc[0][0] = vzero(); acc[0][1] = vzero();
    acc[1][0] = vzero(); acc[1][1] = vzero();

    for (int kc = 0; kc < Cc; kc += 32) {
        v16h a0 = load_frag(orow0 + kc + lh * 8);
        v16h a1 = load_frag(orow1 + kc + lh * 8);
        v16h b0 = load_frag(wrow0 + kc + lh * 8);
        v16h b1 = load_frag(wrow1 + kc + lh * 8);
        acc[0][0] = wmma16(a0, b0, acc[0][0]);
        acc[0][1] = wmma16(a0, b1, acc[0][1]);
        acc[1][0] = wmma16(a1, b0, acc[1][0]);
        acc[1][1] = wmma16(a1, b1, acc[1][1]);
    }

    #pragma unroll
    for (int i = 0; i < 2; ++i) {
        #pragma unroll
        for (int j = 0; j < 2; ++j) {
            #pragma unroll
            for (int r = 0; r < 8; ++r) {
                const int n  = n0 + ns * 32 + i * 16 + r + lh * 8; // D-frag M
                const int co = co0 + cs * 32 + j * 16 + lm;        // D-frag N
                const size_t idx = ((size_t)b * Cc + co) * Nn + n;
                out[idx] = x[idx] + acc[i][j][r] + bo[co];
            }
        }
    }
}

// ---------------------------------------------------------------- launch
extern "C" void kernel_launch(void* const* d_in, const int* in_sizes, int n_in,
                              void* d_out, int out_size, void* d_ws, size_t ws_size,
                              hipStream_t stream) {
    const float* x  = (const float*)d_in[0];
    const float* gs = (const float*)d_in[1];
    const float* gb = (const float*)d_in[2];
    const float* wq = (const float*)d_in[3];
    const float* bq = (const float*)d_in[4];
    const float* wk = (const float*)d_in[5];
    const float* bk = (const float*)d_in[6];
    const float* wv = (const float*)d_in[7];
    const float* bv = (const float*)d_in[8];
    const float* wo = (const float*)d_in[9];
    const float* bo = (const float*)d_in[10];
    float* out = (float*)d_out;

    // workspace layout (f16 elements): total ~44 MB
    half_t* w16 = (half_t*)d_ws;                       // 4 * C * C
    half_t* hT  = w16 + (size_t)4 * Cc * Cc;           // B * N * C
    half_t* qT  = hT  + (size_t)Bb * Nn * Cc;
    half_t* kT  = qT  + (size_t)Bb * Nn * Cc;
    half_t* vB  = kT  + (size_t)Bb * Nn * Cc;
    half_t* oB  = vB  + (size_t)Bb * Nn * Cc;

    k_wcvt<<<dim3(Cc * Cc / 256, 4), 256, 0, stream>>>(wq, wk, wv, wo, w16);
    k_gn  <<<dim3(Bb * 32), 256, 0, stream>>>(x, gs, gb, hT);
    k_qkv <<<dim3(Nn / 64, Cc / 128, Bb * 3), 256, 0, stream>>>(
        w16, hT, bq, bk, bv, qT, kT, vB);
    k_attn<<<dim3(Nn / TQ, Bb), 256, 0, stream>>>(qT, kT, vB, oB);
    k_proj<<<dim3(Nn / 64, Cc / 128, Bb), 256, 0, stream>>>(
        w16 + (size_t)3 * Cc * Cc, oB, x, bo, out);
}
